// BatchFiGNN_91268055040439
// MI455X (gfx1250) — matrix operand order repaired and checked by reading, not compiled, not run on hardware
//
#include <hip/hip_runtime.h>
#include <hip/hip_bf16.h>

typedef __bf16 bf16_t;
typedef __attribute__((ext_vector_type(16))) __bf16 v16bf;
typedef __attribute__((ext_vector_type(8)))  float  v8f;
typedef __attribute__((ext_vector_type(4)))  float  v4f;

#define NB   4096
#define NN   64
#define DD   128
#define OCC  64

union BF16x16 { v16bf v; v4f f[2]; };
union BF16x8  { v4f f; bf16_t b[8]; };

static __device__ inline v8f splat8(float x) {
  v8f r;
#pragma unroll
  for (int i = 0; i < 8; ++i) r[i] = x;
  return r;
}

// A operand: 16x32 bf16 tile from a row-major [rows][ld] array.
// ISA layout: lanes 0-15 row M=lane, v0..3 hold K=(8h)+0..7, v4..7 hold K=16+(8h)+0..7
static __device__ inline v16bf load_A(const bf16_t* base, int ld, int row0, int k0, int lane) {
  int h = lane >> 4, m = lane & 15;
  const bf16_t* p = base + (row0 + m) * ld + k0 + 8 * h;
  BF16x16 u;
  u.f[0] = *reinterpret_cast<const v4f*>(p);
  u.f[1] = *reinterpret_cast<const v4f*>(p + 16);
  return u.v;
}

// B operand: 32x16 bf16 tile; source stored K-contiguous as [ncol][ldk].
// lanes 0-15 col N=lane: v0..3 hold K=k0+(8h)+0..7, v4..7 hold K=k0+16+(8h)+0..7
static __device__ inline v16bf load_B(const bf16_t* base, int ldk, int col0, int k0, int lane) {
  int h = lane >> 4, n = lane & 15;
  const bf16_t* p = base + (col0 + n) * ldk + k0 + 8 * h;
  BF16x16 u;
  u.f[0] = *reinterpret_cast<const v4f*>(p);
  u.f[1] = *reinterpret_cast<const v4f*>(p + 16);
  return u.v;
}

static __device__ inline v8f wmma_bf16(v16bf a, v16bf b, v8f c) {
  return __builtin_amdgcn_wmma_f32_16x16x32_bf16(false, a, false, b, (short)0, c, false, false);
}

static __device__ inline float sigmoidf_fast(float x) {
  return 1.0f / (1.0f + __expf(-x));
}
static __device__ inline float tanhf_fast(float x) {
  // 1 - 2/(e^{2x}+1); saturates correctly at +/-inf, no NaNs
  return 1.0f - 2.0f / (__expf(2.0f * x) + 1.0f);
}

// ---------------- weight prep: fp32 -> bf16, B-operand (K-contiguous) layouts ----------
// ws layout (bf16 elems): wT[128*128] | wih[384*128] | whh[384*128] | m1T[64*128]
__global__ void fignn_prep_kernel(const float* __restrict__ w,
                                  const float* __restrict__ wih_f,
                                  const float* __restrict__ whh_f,
                                  const float* __restrict__ m1_f,
                                  bf16_t* __restrict__ ws) {
  int idx = blockIdx.x * blockDim.x + threadIdx.x;
  bf16_t* wT  = ws;
  bf16_t* wih = ws + 16384;
  bf16_t* whh = ws + 65536;
  bf16_t* m1T = ws + 114688;
  if (idx < 16384) {                       // wT[n][k] = w[k][n]  (B of s@w)
    int n = idx >> 7, k = idx & 127;
    wT[idx] = (bf16_t)w[k * DD + n];
  } else if (idx < 65536) {                // W_ih (3D,D) already [n][k] for a@W_ih^T
    int j = idx - 16384;
    wih[j] = (bf16_t)wih_f[j];
  } else if (idx < 114688) {
    int j = idx - 65536;
    whh[j] = (bf16_t)whh_f[j];
  } else if (idx < 122880) {               // m1T[n][k] = mlp1_w[k][n]
    int j = idx - 114688;
    int n = j >> 7, k = j & 127;
    m1T[j] = (bf16_t)m1_f[k * OCC + n];
  }
}

// ---------------- main kernel: one workgroup per batch element ----------------
__global__ __launch_bounds__(256, 1)
void fignn_kernel(const float* __restrict__ g_h,
                  const float* __restrict__ g_asrc,
                  const float* __restrict__ g_adst,
                  const float* __restrict__ g_bias,
                  const float* __restrict__ g_bih,
                  const float* __restrict__ g_bhh,
                  const float* __restrict__ g_m1b,
                  const float* __restrict__ g_m2w,
                  const float* __restrict__ g_m2b,
                  const bf16_t* __restrict__ wT,
                  const bf16_t* __restrict__ wih,
                  const bf16_t* __restrict__ whh,
                  const bf16_t* __restrict__ m1T,
                  const int* __restrict__ g_steps,
                  float* __restrict__ g_out) {
  __shared__ __attribute__((aligned(16))) float  s_f32[NN * DD];    // master fp32 state
  __shared__ __attribute__((aligned(16))) bf16_t s_bfA[NN * DD];    // bf16 state (A op), dbl-buf
  __shared__ __attribute__((aligned(16))) bf16_t s_bfB[NN * DD];
  __shared__ __attribute__((aligned(16))) bf16_t h_bf [NN * DD];    // residual source
  __shared__ __attribute__((aligned(16))) bf16_t a_bf [NN * DD];    // message tensor (A op)
  __shared__ __attribute__((aligned(16))) bf16_t swT  [DD * NN];    // (s@w)^T, K-contig B op
  __shared__ __attribute__((aligned(16))) bf16_t attn_bf[NN * NN];  // softmax attn (A op)
  __shared__ __attribute__((aligned(16))) float  outp_f32[NN * OCC];// head output staging
  __shared__ float srcv[NN], dstv[NN], wv[NN];

  const int b    = blockIdx.x;
  const int tid  = threadIdx.x;
  const int lane = tid & 31;
  const int wid  = tid >> 5;
  const float* hb = g_h + (size_t)b * NN * DD;

  // ---- load h, init s = h ----
  for (int i = tid; i < NN * DD; i += 256) {
    float v = hb[i];
    s_f32[i] = v;
    bf16_t bv = (bf16_t)v;
    s_bfA[i] = bv;
    h_bf[i]  = bv;
  }
  __syncthreads();

  // ---- attention scores: src[i] = h_i . a_src, dst[i] = h_i . a_dst ----
  if (tid < NN) {
    float ss = 0.f, dd = 0.f;
    for (int d = 0; d < DD; ++d) {
      float hv = s_f32[tid * DD + d];
      ss = fmaf(hv, g_asrc[d], ss);
      dd = fmaf(hv, g_adst[d], dd);
    }
    srcv[tid] = ss;
    dstv[tid] = dd;
  }
  __syncthreads();

  // ---- masked leaky-relu softmax rows; wave handles 8 rows, lane -> cols {lane, lane+32} ----
  for (int q = 0; q < 8; ++q) {
    int r = wid * 8 + q;
    float sr = srcv[r];
    float x0 = sr + dstv[lane];
    x0 = (x0 > 0.f) ? x0 : 0.2f * x0;
    float v0 = (lane == r) ? 0.f : x0;            // mask zeroes diag BEFORE softmax
    float x1 = sr + dstv[lane + 32];
    x1 = (x1 > 0.f) ? x1 : 0.2f * x1;
    float v1 = ((lane + 32) == r) ? 0.f : x1;
    float m = fmaxf(v0, v1);
#pragma unroll
    for (int off = 16; off > 0; off >>= 1) m = fmaxf(m, __shfl_xor(m, off, 32));
    float e0 = __expf(v0 - m), e1 = __expf(v1 - m);
    float s = e0 + e1;
#pragma unroll
    for (int off = 16; off > 0; off >>= 1) s += __shfl_xor(s, off, 32);
    float inv = 1.f / s;
    attn_bf[r * NN + lane]      = (bf16_t)(e0 * inv);
    attn_bf[r * NN + lane + 32] = (bf16_t)(e1 * inv);
  }
  __syncthreads();

  bf16_t* s_cur = s_bfA;
  bf16_t* s_nxt = s_bfB;
  const int steps = g_steps[0];

  for (int it = 0; it < steps; ++it) {
    // ---- pass 1: swT = (s @ w)^T : per wave one column tile, 4 row tiles ----
    for (int t = wid; t < 32; t += 8) {
      int rt = (t >> 3) * 16, ct = (t & 7) * 16;
      v8f acc = splat8(0.f);
#pragma unroll
      for (int k0 = 0; k0 < DD; k0 += 32)
        acc = wmma_bf16(load_A(s_cur, DD, rt, k0, lane), load_B(wT, DD, ct, k0, lane), acc);
      int hh = lane >> 4, n = lane & 15;
      BF16x8 pk;
#pragma unroll
      for (int v = 0; v < 8; ++v) pk.b[v] = (bf16_t)acc[v];
      *reinterpret_cast<v4f*>(&swT[(ct + n) * NN + rt + 8 * hh]) = pk.f;  // transposed, K-contig
    }
    __syncthreads();

    // ---- pass 2: a = attn @ sw + bias ----
    for (int t = wid; t < 32; t += 8) {
      int rt = (t >> 3) * 16, ct = (t & 7) * 16;
      int hh = lane >> 4, n = lane & 15;
      v8f acc = splat8(g_bias[ct + n]);
#pragma unroll
      for (int k0 = 0; k0 < NN; k0 += 32)
        acc = wmma_bf16(load_A(attn_bf, NN, rt, k0, lane), load_B(swT, NN, ct, k0, lane), acc);
#pragma unroll
      for (int v = 0; v < 8; ++v)
        a_bf[(rt + v + 8 * hh) * DD + ct + n] = (bf16_t)acc[v];
    }
    __syncthreads();

    // ---- pass 3: fused GRU — r/z/i_n/h_n accumulated in-register per tile ----
    for (int t = wid; t < 32; t += 8) {
      int rt = (t >> 3) * 16, ct = (t & 7) * 16;
      int hh = lane >> 4, n = lane & 15;
      v8f accr  = splat8(g_bih[ct + n]          + g_bhh[ct + n]);
      v8f accz  = splat8(g_bih[DD + ct + n]     + g_bhh[DD + ct + n]);
      v8f accin = splat8(g_bih[2 * DD + ct + n]);
      v8f acchn = splat8(g_bhh[2 * DD + ct + n]);
#pragma unroll
      for (int k0 = 0; k0 < DD; k0 += 32) {
        v16bf Aa = load_A(a_bf,  DD, rt, k0, lane);
        v16bf As = load_A(s_cur, DD, rt, k0, lane);
        accr  = wmma_bf16(Aa, load_B(wih, DD, ct, k0, lane),          accr);
        accr  = wmma_bf16(As, load_B(whh, DD, ct, k0, lane),          accr);
        accz  = wmma_bf16(Aa, load_B(wih, DD, DD + ct, k0, lane),     accz);
        accz  = wmma_bf16(As, load_B(whh, DD, DD + ct, k0, lane),     accz);
        accin = wmma_bf16(Aa, load_B(wih, DD, 2 * DD + ct, k0, lane), accin);
        acchn = wmma_bf16(As, load_B(whh, DD, 2 * DD + ct, k0, lane), acchn);
      }
#pragma unroll
      for (int v = 0; v < 8; ++v) {
        int row = rt + v + 8 * hh, col = ct + n;
        float r   = sigmoidf_fast(accr[v]);
        float z   = sigmoidf_fast(accz[v]);
        float nn2 = tanhf_fast(accin[v] + r * acchn[v]);
        float sprev = s_f32[row * DD + col];
        float snew  = (1.f - z) * nn2 + z * sprev + (float)h_bf[row * DD + col];
        s_f32[row * DD + col] = snew;
        s_nxt[row * DD + col] = (bf16_t)snew;
      }
    }
    __syncthreads();
    bf16_t* tmp = s_cur; s_cur = s_nxt; s_nxt = tmp;
  }

  // ---- head: outp = s@mlp1 + b ----
  for (int t = wid; t < 16; t += 8) {
    int rt = (t >> 2) * 16, ct = (t & 3) * 16;
    int hh = lane >> 4, n = lane & 15;
    v8f acc = splat8(g_m1b[ct + n]);
#pragma unroll
    for (int k0 = 0; k0 < DD; k0 += 32)
      acc = wmma_bf16(load_A(s_cur, DD, rt, k0, lane), load_B(m1T, DD, ct, k0, lane), acc);
#pragma unroll
    for (int v = 0; v < 8; ++v)
      outp_f32[(rt + v + 8 * hh) * OCC + ct + n] = acc[v];
  }
  // weight[n] = s_n . mlp2_w + b
  if (tid < NN) {
    float acc = g_m2b[0];
    for (int d = 0; d < DD; ++d) acc = fmaf(s_f32[tid * DD + d], g_m2w[d], acc);
    wv[tid] = acc;
  }
  __syncthreads();
  // out[c] = sum_n wv[n] * outp[n][c]
  if (tid < OCC) {
    float acc = 0.f;
    for (int n2 = 0; n2 < NN; ++n2) acc = fmaf(wv[n2], outp_f32[n2 * OCC + tid], acc);
    g_out[(size_t)b * OCC + tid] = acc;
  }
}

extern "C" void kernel_launch(void* const* d_in, const int* in_sizes, int n_in,
                              void* d_out, int out_size, void* d_ws, size_t ws_size,
                              hipStream_t stream) {
  const float* h    = (const float*)d_in[0];
  /* d_in[1] = adj : unused by the reference math (shape only) */
  const float* asrc = (const float*)d_in[2];
  const float* adst = (const float*)d_in[3];
  const float* w    = (const float*)d_in[4];
  const float* bias = (const float*)d_in[5];
  const float* Wih  = (const float*)d_in[6];
  const float* Whh  = (const float*)d_in[7];
  const float* bih  = (const float*)d_in[8];
  const float* bhh  = (const float*)d_in[9];
  const float* m1w  = (const float*)d_in[10];
  const float* m1b  = (const float*)d_in[11];
  const float* m2w  = (const float*)d_in[12];
  const float* m2b  = (const float*)d_in[13];
  const int* steps  = (const int*)d_in[14];
  float* out  = (float*)d_out;
  bf16_t* ws  = (bf16_t*)d_ws;

  fignn_prep_kernel<<<480, 256, 0, stream>>>(w, Wih, Whh, m1w, ws);
  fignn_kernel<<<NB, 256, 0, stream>>>(h, asrc, adst, bias, bih, bhh, m1b, m2w, m2b,
                                       ws, ws + 16384, ws + 65536, ws + 114688,
                                       steps, out);
}
